// MultiHeadAttention_790273982899
// MI455X (gfx1250) — matrix operand order repaired and checked by reading, not compiled, not run on hardware
//
#include <hip/hip_runtime.h>
#include <hip/hip_bf16.h>

// MHA: B=16, H=96, C=184, E=256, HEADS=8, RD=32.  N = B*C = 2944.
// fp32 -> bf16 (native v_cvt), all GEMMs via v_wmma_f32_16x16x32_bf16,
// softmax in fp32.  V stored transposed for contiguous-K B-fragments.
// Proj kernels stage the shared A-tile through LDS once per block.
// Attention stages K/Vt tiles into LDS via GLOBAL_LOAD_ASYNC_TO_LDS_B128
// (ASYNCcnt) when available, shared by all 6 waves.
// Workspace use: ~580 MB (bf16 weights + Q + K + Vt + Ctx).

#define Bn   16
#define Hh   96
#define Cc   184
#define Ee   256
#define HEADS 8
#define RD   32
#define Nn   (Bn*Cc)            // 2944
#define Mrows (Nn*Hh)           // 282624
#define MTILES (Mrows/16)       // 17664
#define ROWSTRIDE (Cc*Ee)       // 47104 floats between consecutive h in input
#define SCALE 0.17677669529663687f  // 1/sqrt(32)
#define APAD 264                // proj LDS A row stride (bf16): bank stagger
#define KPAD 40                 // attn LDS K row stride  (20 dwords)
#define VPAD 104                // attn LDS Vt row stride (52 dwords)

typedef __bf16 bf16_t;
typedef bf16_t bf16x8  __attribute__((ext_vector_type(8)));
typedef bf16_t bf16x16 __attribute__((ext_vector_type(16)));
typedef float  v8f     __attribute__((ext_vector_type(8)));
typedef float  v4f     __attribute__((ext_vector_type(4)));
typedef int    i32x4   __attribute__((ext_vector_type(4)));

// ---------- async copy global->LDS (CDNA5), with compile-safe fallback ------
#if defined(__has_builtin)
#if __has_builtin(__builtin_amdgcn_global_load_async_to_lds_b128)
#define HAVE_ASYNC_LDS 1
#endif
#endif

#ifdef HAVE_ASYNC_LDS
__device__ __forceinline__ void cp16_g2l(const void* g, void* l) {
  // builtin expects pointers to int __vector(4) (per hipcc diagnostic)
  __builtin_amdgcn_global_load_async_to_lds_b128((i32x4*)g, (i32x4*)l, 0, 0);
}
#if __has_builtin(__builtin_amdgcn_s_wait_asynccnt)
__device__ __forceinline__ void wait_async0() { __builtin_amdgcn_s_wait_asynccnt(0); }
#else
__device__ __forceinline__ void wait_async0() { asm volatile("s_wait_asynccnt 0" ::: "memory"); }
#endif
#else
__device__ __forceinline__ void cp16_g2l(const void* g, void* l) {
  *(bf16x8*)l = *(const bf16x8*)g;
}
__device__ __forceinline__ void wait_async0() {}
#endif

// ---------- helpers ----------
__device__ __forceinline__ bf16_t f2bf(float f) { return (bf16_t)f; }

__device__ __forceinline__ bf16x16 cat8(bf16x8 lo, bf16x8 hi) {
  return __builtin_shufflevector(lo, hi, 0,1,2,3,4,5,6,7,8,9,10,11,12,13,14,15);
}

__device__ __forceinline__ bf16x8 cvt8(v4f a, v4f b) {
  bf16x8 r;
  r[0]=f2bf(a[0]); r[1]=f2bf(a[1]); r[2]=f2bf(a[2]); r[3]=f2bf(a[3]);
  r[4]=f2bf(b[0]); r[5]=f2bf(b[1]); r[6]=f2bf(b[2]); r[7]=f2bf(b[3]);
  return r;
}

// A-fragment 16x32 bf16: lanes 0-15 -> K {sw..sw+7, 16+sw..+7}, sw=(lane>=16)*8
__device__ __forceinline__ bf16x16 load_a_bf16(const bf16_t* p, int sw) {
  bf16x8 lo = *(const bf16x8*)(p + sw);
  bf16x8 hi = *(const bf16x8*)(p + 16 + sw);
  return cat8(lo, hi);
}
// B-fragment 32x16 bf16: lane holds column (lane&15), K {k16/2..+7, 16+k16/2..+7}
__device__ __forceinline__ bf16x16 load_b_bf16(const bf16_t* p, int k16) {
  bf16x8 lo = *(const bf16x8*)(p + k16/2);
  bf16x8 hi = *(const bf16x8*)(p + 16 + k16/2);
  return cat8(lo, hi);
}

__device__ __forceinline__ v8f wmma_bf16(bf16x16 a, bf16x16 b, v8f c) {
  return __builtin_amdgcn_wmma_f32_16x16x32_bf16(false, a, false, b, (short)0, c, false, false);
}

// ---------- kernel 0: fp32 weights -> bf16 (wq|wk|wv|wc concatenated) ----------
__global__ __launch_bounds__(256) void cvt_weights_kernel(
    const float* __restrict__ wq, const float* __restrict__ wk,
    const float* __restrict__ wv, const float* __restrict__ wc,
    bf16_t* __restrict__ dst)
{
  int i = blockIdx.x * 256 + threadIdx.x;          // 0 .. 262143
  const float* srcs[4] = {wq, wk, wv, wc};
  dst[i] = f2bf(srcs[i >> 16][i & 65535]);
}

// ---------- kernel 1: projection  Y = X @ W^T  (per-wave 16x32 tile) ----------
__global__ __launch_bounds__(256) void proj_kernel(
    const float* __restrict__ X, const bf16_t* __restrict__ Wb,
    bf16_t* __restrict__ Dst, int transposed)
{
  __shared__ __align__(16) bf16_t Alds[16][APAD];
  const int tid  = threadIdx.x;
  const int lane = tid & 31;
  const int wave = tid >> 5;               // 0..7
  const int ln   = lane & 15;
  const int hi   = lane >> 4;              // 0/1
  const int row0 = blockIdx.x * 16;        // M-tile (never crosses n: 96%16==0)
  const int n    = row0 / Hh;
  const int h0   = row0 % Hh;
  const int b    = n / Cc;
  const int c    = n % Cc;
  const int o0   = wave * 32;

  // cooperative load + convert of the shared A-tile (16 rows x 256 f32 -> bf16)
  {
    const int rrow = tid >> 4;             // 0..15
    const int kseg = (tid & 15) * 16;      // 0,16,...,240
    const float* src = X + ((size_t)((b*Hh + h0 + rrow))*Cc + c)*Ee + kseg;
    v4f f0 = *(const v4f*)(src);
    v4f f1 = *(const v4f*)(src + 4);
    v4f f2 = *(const v4f*)(src + 8);
    v4f f3 = *(const v4f*)(src + 12);
    *(bf16x8*)(&Alds[rrow][kseg])     = cvt8(f0, f1);
    *(bf16x8*)(&Alds[rrow][kseg + 8]) = cvt8(f2, f3);
  }
  __syncthreads();

  const bf16_t* w0 = Wb + (size_t)(o0 + ln) * Ee;
  const bf16_t* w1 = Wb + (size_t)(o0 + 16 + ln) * Ee;

  v8f acc0 = {}; v8f acc1 = {};
  #pragma unroll
  for (int k0 = 0; k0 < Ee; k0 += 32) {
    bf16x16 a = cat8(*(const bf16x8*)(&Alds[ln][k0 + hi*8]),
                     *(const bf16x8*)(&Alds[ln][k0 + 16 + hi*8]));
    bf16x16 b0 = load_b_bf16(w0 + k0, hi*16);
    bf16x16 b1 = load_b_bf16(w1 + k0, hi*16);
    acc0 = wmma_bf16(a, b0, acc0);
    acc1 = wmma_bf16(a, b1, acc1);
  }

  if (!transposed) {
    const int mrow = row0 + hi*8;          // C-fragment: vgpr r -> row m = r + 8*hi
    #pragma unroll
    for (int r = 0; r < 8; ++r) {
      Dst[(size_t)(mrow + r)*Ee + o0 + ln]      = f2bf(acc0[r]);
      Dst[(size_t)(mrow + r)*Ee + o0 + 16 + ln] = f2bf(acc1[r]);
    }
  } else {
    // lane owns one output column o; 8 consecutive h values -> packed b128 store
    const int hbase = h0 + hi*8;
    bf16x8 p0, p1;
    #pragma unroll
    for (int r = 0; r < 8; ++r) { p0[r] = f2bf(acc0[r]); p1[r] = f2bf(acc1[r]); }
    *(bf16x8*)(Dst + ((size_t)n*Ee + o0 + ln)*Hh + hbase)      = p0;
    *(bf16x8*)(Dst + ((size_t)n*Ee + o0 + 16 + ln)*Hh + hbase) = p1;
  }
}

// ---------- kernel 2: attention per (n, head); 6 waves x 16 query rows ----------
__global__ __launch_bounds__(192) void attn_kernel(
    const bf16_t* __restrict__ Q, const bf16_t* __restrict__ K,
    const bf16_t* __restrict__ Vt, bf16_t* __restrict__ Ctx)
{
  __shared__ __align__(16) bf16_t Klds[Hh][KPAD];    // 96 x 32 used
  __shared__ __align__(16) bf16_t Vlds[RD][VPAD];    // 32 x 96 used
  __shared__ __align__(16) bf16_t Plds[6][16][104];  // per-wave P staging
  const int tid  = threadIdx.x;
  const int lane = tid & 31;
  const int wave = tid >> 5;       // 0..5
  const int ln   = lane & 15;
  const int hi   = lane >> 4;
  const int n    = blockIdx.x >> 3;
  const int head = blockIdx.x & 7;
  const int q0   = wave * 16;

  const bf16_t* Qb = Q + (size_t)n*Hh*Ee + head*RD;
  const bf16_t* Kb = K + (size_t)n*Hh*Ee + head*RD;
  const bf16_t* Vb = Vt + ((size_t)n*Ee + head*RD) * Hh;

  // cooperative async staging of K (96x32) and Vt (32x96) tiles into LDS
  #pragma unroll
  for (int it = 0; it < 2; ++it) {
    int idx = tid + it*192;                // 0..383 (16B chunks)
    int krow = idx >> 2, kch = idx & 3;    // K: 96 rows x 4 chunks
    cp16_g2l(Kb + (size_t)krow*Ee + kch*8, &Klds[krow][kch*8]);
    int vrow = idx / 12, vch = idx % 12;   // Vt: 32 rows x 12 chunks
    cp16_g2l(Vb + (size_t)vrow*Hh + vch*8, &Vlds[vrow][vch*8]);
  }
  wait_async0();
  __syncthreads();

  // gemm1: S[16 x 96] = Q[16 x 32] * K^T   (single K-step: RD == 32)
  bf16x16 a = load_a_bf16(Qb + (size_t)(q0 + ln)*Ee, hi*8);
  v8f S[6];
  #pragma unroll
  for (int jt = 0; jt < 6; ++jt) {
    bf16x16 bb = load_b_bf16(&Klds[jt*16 + ln][0], hi*16);
    v8f z = {};
    S[jt] = wmma_bf16(a, bb, z);
  }

  // softmax over 96 keys: reduce across 6 tiles + 16 lanes (halves separate)
  float rcp[8];
  #pragma unroll
  for (int r = 0; r < 8; ++r) {
    float mx = S[0][r];
    #pragma unroll
    for (int jt = 1; jt < 6; ++jt) mx = fmaxf(mx, S[jt][r]);
    #pragma unroll
    for (int off = 1; off < 16; off <<= 1) mx = fmaxf(mx, __shfl_xor(mx, off, 32));
    float sum = 0.f;
    #pragma unroll
    for (int jt = 0; jt < 6; ++jt) {
      float p = __expf((S[jt][r] - mx) * SCALE);
      S[jt][r] = p;
      sum += p;
    }
    #pragma unroll
    for (int off = 1; off < 16; off <<= 1) sum += __shfl_xor(sum, off, 32);
    rcp[r] = 1.0f / sum;
  }

  // stage P (C-fragment layout) into LDS row-major to re-load as A-fragment
  #pragma unroll
  for (int jt = 0; jt < 6; ++jt)
    #pragma unroll
    for (int r = 0; r < 8; ++r)
      Plds[wave][r + hi*8][jt*16 + ln] = f2bf(S[jt][r]);

  // gemm2: O[16 x 32] = P[16 x 96] * V[96 x 32]  (Vt rows = contiguous key idx)
  v8f O0 = {}, O1 = {};
  #pragma unroll
  for (int kc = 0; kc < 3; ++kc) {
    const bf16_t* ap = &Plds[wave][ln][kc*32 + hi*8];
    bf16x16 pa = cat8(*(const bf16x8*)ap, *(const bf16x8*)(ap + 16));
    bf16x16 b0 = load_b_bf16(&Vlds[ln][kc*32],      hi*16);
    bf16x16 b1 = load_b_bf16(&Vlds[16 + ln][kc*32], hi*16);
    O0 = wmma_bf16(pa, b0, O0);
    O1 = wmma_bf16(pa, b1, O1);
  }

  // normalize and store merged-head context [N, H, E] bf16
  const size_t crow = ((size_t)n*Hh + q0 + hi*8)*Ee + head*RD;
  #pragma unroll
  for (int r = 0; r < 8; ++r) {
    Ctx[crow + (size_t)r*Ee + ln]      = f2bf(O0[r] * rcp[r]);
    Ctx[crow + (size_t)r*Ee + 16 + ln] = f2bf(O1[r] * rcp[r]);
  }
}

// ---------- kernel 3: out = Ctx @ Wc^T + b, scatter to [B, H, C, E] fp32 ----------
__global__ __launch_bounds__(256) void outproj_kernel(
    const bf16_t* __restrict__ Ctx, const bf16_t* __restrict__ Wc,
    const float* __restrict__ bias, float* __restrict__ Out)
{
  const int lane = threadIdx.x & 31;
  const int wave = threadIdx.x >> 5;
  const int ln   = lane & 15;
  const int hi   = lane >> 4;
  const int row0 = blockIdx.x * 16;
  const int n    = row0 / Hh;
  const int h0   = row0 % Hh;
  const int b    = n / Cc;
  const int c    = n % Cc;
  const int o0   = wave * 32;

  const bf16_t* arow = Ctx + (size_t)(row0 + ln) * Ee;
  const bf16_t* w0   = Wc + (size_t)(o0 + ln) * Ee;
  const bf16_t* w1   = Wc + (size_t)(o0 + 16 + ln) * Ee;

  v8f acc0 = {}; v8f acc1 = {};
  #pragma unroll
  for (int k0 = 0; k0 < Ee; k0 += 32) {
    bf16x16 aa = load_a_bf16(arow + k0, hi*8);
    acc0 = wmma_bf16(aa, load_b_bf16(w0 + k0, hi*16), acc0);
    acc1 = wmma_bf16(aa, load_b_bf16(w1 + k0, hi*16), acc1);
  }

  const float bias0 = bias[o0 + ln];
  const float bias1 = bias[o0 + 16 + ln];
  const size_t obase = ((size_t)(b*Hh + h0 + hi*8)*Cc + c) * Ee;
  #pragma unroll
  for (int r = 0; r < 8; ++r) {
    Out[obase + (size_t)r*ROWSTRIDE + o0 + ln]      = acc0[r] + bias0;
    Out[obase + (size_t)r*ROWSTRIDE + o0 + 16 + ln] = acc1[r] + bias1;
  }
}

// ---------- launcher ----------
extern "C" void kernel_launch(void* const* d_in, const int* in_sizes, int n_in,
                              void* d_out, int out_size, void* d_ws, size_t ws_size,
                              hipStream_t stream) {
  (void)in_sizes; (void)n_in; (void)out_size; (void)ws_size;
  const float* values  = (const float*)d_in[0];
  const float* queries = (const float*)d_in[1];
  const float* keys    = (const float*)d_in[2];
  const float* w_q     = (const float*)d_in[3];
  const float* w_k     = (const float*)d_in[4];
  const float* w_v     = (const float*)d_in[5];
  const float* w_c     = (const float*)d_in[6];
  const float* b_c     = (const float*)d_in[7];
  float* out = (float*)d_out;

  // workspace layout (bf16 elements)
  bf16_t* wbf = (bf16_t*)d_ws;                 // 4 x 65536 bf16 weights
  const size_t BIG = (size_t)Nn * Hh * Ee;     // 72,351,744 elems per tensor
  bf16_t* Qb  = wbf + 262144;
  bf16_t* Kb  = Qb + BIG;
  bf16_t* Vt  = Kb + BIG;
  bf16_t* Ctx = Vt + BIG;

  cvt_weights_kernel<<<1024, 256, 0, stream>>>(w_q, w_k, w_v, w_c, wbf);
  proj_kernel<<<MTILES, 256, 0, stream>>>(queries, wbf,          Qb, 0);
  proj_kernel<<<MTILES, 256, 0, stream>>>(keys,    wbf + 65536,  Kb, 0);
  proj_kernel<<<MTILES, 256, 0, stream>>>(values,  wbf + 131072, Vt, 1);
  attn_kernel<<<Nn * HEADS, 192, 0, stream>>>(Qb, Kb, Vt, Ctx);
  outproj_kernel<<<MTILES, 256, 0, stream>>>(Ctx, wbf + 196608, b_c, out);
}
